// MHA_81561428951706
// MI455X (gfx1250) — matrix operand order repaired and checked by reading, not compiled, not run on hardware
//
#include <hip/hip_runtime.h>
#include <hip/hip_bf16.h>

// ---------------------------------------------------------------------------
// MHA for MI455X (gfx1250): bf16 WMMA pipeline, flash-style attention.
// Round 3: ping-pong double buffering (no register rotation -> no WAR-hazard
// v_nops), flash j-loop unrolled x2 for cross-iteration load hoisting.
// ---------------------------------------------------------------------------

#define EMBED    1024
#define HEADS    16
#define HEAD_DIM 64
#define BATCH    2
#define SEQ      2048
#define ROWS     (BATCH * SEQ)   // 4096 token rows

typedef __attribute__((ext_vector_type(16))) __bf16 bf16x16;
typedef __attribute__((ext_vector_type(8)))  __bf16 bf16x8;
typedef __attribute__((ext_vector_type(8)))  float  f32x8;

// ---- WMMA fragment loaders (wave32 layouts per CDNA5 ISA 7.12.2) ----------

// A-matrix 16x32 bf16: lane L -> row m=L&15; element chunks k in
// [(L>>4)*8, +8) and [16+(L>>4)*8, +8). Two contiguous 16B loads.
__device__ __forceinline__ bf16x16 load_frag_a(const __bf16* base, int ld,
                                               int row0, int k0) {
  const int lane = threadIdx.x & 31;
  const int m    = lane & 15;
  const int kh   = (lane >> 4) << 3;   // 0 or 8
  const __bf16* p = base + (size_t)(row0 + m) * ld + k0;
  bf16x8 lo = *(const bf16x8*)(p + kh);
  bf16x8 hi = *(const bf16x8*)(p + 16 + kh);
  bf16x16 a;
#pragma unroll
  for (int i = 0; i < 8; ++i) { a[i] = lo[i]; a[i + 8] = hi[i]; }
  return a;
}

// B-matrix 32x16 bf16: lane L -> column n=n0+(L&15); element chunks k in
// [(L>>4)*8, +8) and [16+(L>>4)*8, +8) (strided loads).
__device__ __forceinline__ bf16x16 load_frag_b(const __bf16* base, int ld,
                                               int k0, int n0) {
  const int lane = threadIdx.x & 31;
  const int kh   = (lane >> 4) << 3;
  const __bf16* p = base + (size_t)(k0 + kh) * ld + n0 + (lane & 15);
  bf16x16 b;
#pragma unroll
  for (int i = 0; i < 8; ++i) { b[i] = p[(size_t)i * ld]; b[i + 8] = p[(size_t)(16 + i) * ld]; }
  return b;
}

#define WMMA_BF16(A, B, C) \
  __builtin_amdgcn_wmma_f32_16x16x32_bf16(false, (A), false, (B), (short)0, (C), false, false)

// Reductions across the 16 lanes of a half-wave (masks < 16 never cross the
// lane<16 / lane>=16 boundary, matching the C/D row replication).
__device__ __forceinline__ float half_max(float v) {
#pragma unroll
  for (int m = 1; m < 16; m <<= 1) v = fmaxf(v, __shfl_xor(v, m, 32));
  return v;
}
__device__ __forceinline__ float half_sum(float v) {
#pragma unroll
  for (int m = 1; m < 16; m <<= 1) v += __shfl_xor(v, m, 32);
  return v;
}

// ---- Prep kernels ---------------------------------------------------------

__global__ void convert_x_kernel(const float* __restrict__ x,
                                 __bf16* __restrict__ xb, int n) {
  int i = blockIdx.x * blockDim.x + threadIdx.x;
  if (i < n) xb[i] = (__bf16)x[i];
}

// Wt[z][k][j] = (bf16) W_z[j][k]   (32x32 LDS tile transpose)
__global__ __launch_bounds__(1024)
void transpose_w_kernel(const float* __restrict__ W0, const float* __restrict__ W1,
                        const float* __restrict__ W2, const float* __restrict__ W3,
                        __bf16* __restrict__ Wt) {
  __shared__ float tile[32][33];
  const float* W = (blockIdx.z == 0) ? W0 : (blockIdx.z == 1) ? W1
                 : (blockIdx.z == 2) ? W2 : W3;
  __bf16* out = Wt + (size_t)blockIdx.z * EMBED * EMBED;
  int j = blockIdx.y * 32 + threadIdx.y;   // row of W
  int k = blockIdx.x * 32 + threadIdx.x;   // col of W
  tile[threadIdx.y][threadIdx.x] = W[(size_t)j * EMBED + k];
  __syncthreads();
  int ko = blockIdx.x * 32 + threadIdx.y;
  int jo = blockIdx.y * 32 + threadIdx.x;
  out[(size_t)ko * EMBED + jo] = (__bf16)tile[threadIdx.x][threadIdx.y];
}

// ---- Projection GEMM: Y = Xb @ Wt + bias ----------------------------------
// Each wave owns a 32x64 tile (8 WMMAs per 32-deep K-step). K-loop is
// unrolled x2 with two *named* fragment sets ping-ponging: compute on set 0
// while set 1 loads, then swap. No register rotation -> no WMMA-source WAR
// hazard NOPs, and loadcnt waits always cover loads issued a full step early.
// mode 0: Q -> [b,h,n,d] bf16, prescaled by EMBED^-0.5
// mode 1: K -> [b,h,d,n] bf16
// mode 2: V -> [b,h,n,d] bf16
// mode 3: fp32 row-major -> d_out
__global__ __launch_bounds__(256)
void gemm_bias_kernel(const __bf16* __restrict__ X, const __bf16* __restrict__ Wt,
                      const float* __restrict__ bias, __bf16* __restrict__ out_bf,
                      float* __restrict__ out_f, int mode) {
  const int wave = threadIdx.x >> 5;
  const int lane = threadIdx.x & 31;
  const int row0 = blockIdx.x * 256 + wave * 32;
  const int col0 = blockIdx.y * 64;

  f32x8 acc[2][4] = {};

  // Fragment set 0, preloaded for k = 0.
  bf16x16 p0a0 = load_frag_a(X, EMBED, row0, 0);
  bf16x16 p0a1 = load_frag_a(X, EMBED, row0 + 16, 0);
  bf16x16 p0b0 = load_frag_b(Wt, EMBED, 0, col0);
  bf16x16 p0b1 = load_frag_b(Wt, EMBED, 0, col0 + 16);
  bf16x16 p0b2 = load_frag_b(Wt, EMBED, 0, col0 + 32);
  bf16x16 p0b3 = load_frag_b(Wt, EMBED, 0, col0 + 48);

  for (int k = 0; k < EMBED; k += 64) {
    // Load set 1 for k+32 (always in range: EMBED is a multiple of 64).
    const bf16x16 p1a0 = load_frag_a(X, EMBED, row0, k + 32);
    const bf16x16 p1a1 = load_frag_a(X, EMBED, row0 + 16, k + 32);
    const bf16x16 p1b0 = load_frag_b(Wt, EMBED, k + 32, col0);
    const bf16x16 p1b1 = load_frag_b(Wt, EMBED, k + 32, col0 + 16);
    const bf16x16 p1b2 = load_frag_b(Wt, EMBED, k + 32, col0 + 32);
    const bf16x16 p1b3 = load_frag_b(Wt, EMBED, k + 32, col0 + 48);

    // Compute on set 0 (k) while set 1 loads are in flight.
    acc[0][0] = WMMA_BF16(p0a0, p0b0, acc[0][0]);
    acc[0][1] = WMMA_BF16(p0a0, p0b1, acc[0][1]);
    acc[0][2] = WMMA_BF16(p0a0, p0b2, acc[0][2]);
    acc[0][3] = WMMA_BF16(p0a0, p0b3, acc[0][3]);
    acc[1][0] = WMMA_BF16(p0a1, p0b0, acc[1][0]);
    acc[1][1] = WMMA_BF16(p0a1, p0b1, acc[1][1]);
    acc[1][2] = WMMA_BF16(p0a1, p0b2, acc[1][2]);
    acc[1][3] = WMMA_BF16(p0a1, p0b3, acc[1][3]);

    // Reload set 0 for k+64 (skipped on the final step).
    if (k + 64 < EMBED) {
      p0a0 = load_frag_a(X, EMBED, row0, k + 64);
      p0a1 = load_frag_a(X, EMBED, row0 + 16, k + 64);
      p0b0 = load_frag_b(Wt, EMBED, k + 64, col0);
      p0b1 = load_frag_b(Wt, EMBED, k + 64, col0 + 16);
      p0b2 = load_frag_b(Wt, EMBED, k + 64, col0 + 32);
      p0b3 = load_frag_b(Wt, EMBED, k + 64, col0 + 48);
    }

    // Compute on set 1 (k+32) while set 0 loads are in flight.
    acc[0][0] = WMMA_BF16(p1a0, p1b0, acc[0][0]);
    acc[0][1] = WMMA_BF16(p1a0, p1b1, acc[0][1]);
    acc[0][2] = WMMA_BF16(p1a0, p1b2, acc[0][2]);
    acc[0][3] = WMMA_BF16(p1a0, p1b3, acc[0][3]);
    acc[1][0] = WMMA_BF16(p1a1, p1b0, acc[1][0]);
    acc[1][1] = WMMA_BF16(p1a1, p1b1, acc[1][1]);
    acc[1][2] = WMMA_BF16(p1a1, p1b2, acc[1][2]);
    acc[1][3] = WMMA_BF16(p1a1, p1b3, acc[1][3]);
  }

  const int mrow = (lane >> 4) << 3;
  const int nl   = lane & 15;
  const float osc = (mode == 0) ? 0.03125f : 1.0f;   // fold EMBED^-0.5 into Q
#pragma unroll
  for (int r = 0; r < 2; ++r) {
#pragma unroll
    for (int t = 0; t < 4; ++t) {
      const int col = col0 + 16 * t + nl;
      const float bval = bias[col];
      const int h = col >> 6, d = col & 63;
#pragma unroll
      for (int v = 0; v < 8; ++v) {
        const int row = row0 + 16 * r + mrow + v;
        const int bb = row >> 11, n = row & (SEQ - 1);
        const float val = (acc[r][t][v] + bval) * osc;
        if (mode == 0 || mode == 2) {          // [b,h,n,d]
          out_bf[((size_t)((bb * HEADS + h) * SEQ + n)) * HEAD_DIM + d] = (__bf16)val;
        } else if (mode == 1) {                // [b,h,d,n]
          out_bf[((size_t)((bb * HEADS + h) * HEAD_DIM + d)) * SEQ + n] = (__bf16)val;
        } else {                               // fp32 row-major output
          out_f[(size_t)row * EMBED + col] = val;
        }
      }
    }
  }
}

// ---- Flash attention: 1 wave = one (b,h, 16-row) query block --------------
__global__ __launch_bounds__(128)
void flash_attn_kernel(const __bf16* __restrict__ Q, const __bf16* __restrict__ K,
                       const __bf16* __restrict__ V, __bf16* __restrict__ O) {
  __shared__ __bf16 psmem[4][16 * 40];       // per-wave 16x32 P tile, padded rows
  const int wave = threadIdx.x >> 5;
  const int lane = threadIdx.x & 31;
  const int wid  = blockIdx.x * 4 + wave;    // 0..4095
  const int ib   = wid & 127;
  const int h    = (wid >> 7) & 15;
  const int b    = wid >> 11;
  const int i0   = ib * 16;

  const __bf16* Qh = Q + ((size_t)(b * HEADS + h) * SEQ) * HEAD_DIM;
  const __bf16* Kh = K + ((size_t)(b * HEADS + h) * HEAD_DIM) * SEQ;  // [d][n]
  const __bf16* Vh = V + ((size_t)(b * HEADS + h) * SEQ) * HEAD_DIM;  // [n][d]
  __bf16* pbuf = &psmem[wave][0];

  // Query fragments (prescaled by EMBED^-0.5 in the Q projection) stay in
  // registers for the whole sweep (d = 64 -> 2 frags).
  const bf16x16 qa0 = load_frag_a(Qh, HEAD_DIM, i0, 0);
  const bf16x16 qa1 = load_frag_a(Qh, HEAD_DIM, i0, 32);

  f32x8 oacc[4] = {};
  float rm[8], rl[8];
#pragma unroll
  for (int v = 0; v < 8; ++v) { rm[v] = -1e30f; rl[v] = 0.0f; }

  const int mrow = (lane >> 4) << 3;
  const int nl   = lane & 15;

#pragma unroll 2   // let the scheduler hoist step j+1 loads over step j's tail
  for (int j0 = 0; j0 < SEQ; j0 += 32) {
    // ---- issue ALL of this step's global loads up front: K frags feed the
    // score WMMAs; V frags are only needed after softmax, so their latency
    // hides behind 4 WMMAs + the whole softmax VALU/exp chunk. ----
    const bf16x16 kb0 = load_frag_b(Kh, SEQ, 0,  j0);
    const bf16x16 kb1 = load_frag_b(Kh, SEQ, 32, j0);
    const bf16x16 kb2 = load_frag_b(Kh, SEQ, 0,  j0 + 16);
    const bf16x16 kb3 = load_frag_b(Kh, SEQ, 32, j0 + 16);
    const bf16x16 vb0 = load_frag_b(Vh, HEAD_DIM, j0, 0);
    const bf16x16 vb1 = load_frag_b(Vh, HEAD_DIM, j0, 16);
    const bf16x16 vb2 = load_frag_b(Vh, HEAD_DIM, j0, 32);
    const bf16x16 vb3 = load_frag_b(Vh, HEAD_DIM, j0, 48);

    // ---- scores S = (Q*scale) K^T for columns [j0, j0+32) ----
    f32x8 s0 = {}, s1 = {};
    s0 = WMMA_BF16(qa0, kb0, s0);
    s0 = WMMA_BF16(qa1, kb1, s0);
    s1 = WMMA_BF16(qa0, kb2, s1);
    s1 = WMMA_BF16(qa1, kb3, s1);

    // ---- online softmax (per row m = mrow+v, replicated across half) ----
    float corr[8];
#pragma unroll
    for (int v = 0; v < 8; ++v) {
      const float a = s0[v];
      const float c = s1[v];
      const float tmax = half_max(fmaxf(a, c));
      const float mnew = fmaxf(rm[v], tmax);
      corr[v] = __expf(rm[v] - mnew);
      const float p0 = __expf(a - mnew);
      const float p1 = __expf(c - mnew);
      rl[v] = rl[v] * corr[v] + half_sum(p0 + p1);
      rm[v] = mnew;
      // C-layout -> LDS (row m, cols j0..j0+31), re-read below in A-layout.
      pbuf[(mrow + v) * 40 + nl]      = (__bf16)p0;
      pbuf[(mrow + v) * 40 + 16 + nl] = (__bf16)p1;
    }
#pragma unroll
    for (int t = 0; t < 4; ++t)
#pragma unroll
      for (int v = 0; v < 8; ++v) oacc[t][v] *= corr[v];

    // ---- O += P @ V  (P: 16x32 via LDS transpose; V already in registers) --
    const bf16x16 pa = load_frag_a(pbuf, 40, 0, 0);
    oacc[0] = WMMA_BF16(pa, vb0, oacc[0]);
    oacc[1] = WMMA_BF16(pa, vb1, oacc[1]);
    oacc[2] = WMMA_BF16(pa, vb2, oacc[2]);
    oacc[3] = WMMA_BF16(pa, vb3, oacc[3]);
  }

  // ---- finalize: divide by row sums, emit [b, n, h*64+d] bf16 ----
  float inv[8];
#pragma unroll
  for (int v = 0; v < 8; ++v) inv[v] = 1.0f / rl[v];
#pragma unroll
  for (int t = 0; t < 4; ++t)
#pragma unroll
    for (int v = 0; v < 8; ++v) {
      const int row = i0 + mrow + v;
      O[((size_t)(b * SEQ + row)) * EMBED + h * HEAD_DIM + 16 * t + nl] =
          (__bf16)(oacc[t][v] * inv[v]);
    }
}

// ---------------------------------------------------------------------------
extern "C" void kernel_launch(void* const* d_in, const int* in_sizes, int n_in,
                              void* d_out, int out_size, void* d_ws, size_t ws_size,
                              hipStream_t stream) {
  const float* x  = (const float*)d_in[0];
  const float* Wq = (const float*)d_in[1];
  const float* bq = (const float*)d_in[2];
  const float* Wk = (const float*)d_in[3];
  const float* bk = (const float*)d_in[4];
  const float* Wv = (const float*)d_in[5];
  const float* bv = (const float*)d_in[6];
  const float* Wo = (const float*)d_in[7];
  const float* bo = (const float*)d_in[8];
  float* out = (float*)d_out;

  // Workspace layout (40 MB total):
  //   [0,8MB)   xb: bf16(x); reused as attention output after QKV projections
  //   [8,16MB)  Wt: 4 transposed bf16 weights (Wq,Wk,Wv,Wo)
  //   [16,24MB) Q  [b,h,n,d] bf16 (prescaled by EMBED^-0.5)
  //   [24,32MB) K  [b,h,d,n] bf16
  //   [32,40MB) V  [b,h,n,d] bf16
  char* ws = (char*)d_ws;
  __bf16* xb  = (__bf16*)(ws);
  __bf16* Wt  = (__bf16*)(ws + ((size_t)8  << 20));
  __bf16* Qb  = (__bf16*)(ws + ((size_t)16 << 20));
  __bf16* Kb  = (__bf16*)(ws + ((size_t)24 << 20));
  __bf16* Vb  = (__bf16*)(ws + ((size_t)32 << 20));
  __bf16* Wqt = Wt + (size_t)0 * EMBED * EMBED;
  __bf16* Wkt = Wt + (size_t)1 * EMBED * EMBED;
  __bf16* Wvt = Wt + (size_t)2 * EMBED * EMBED;
  __bf16* Wot = Wt + (size_t)3 * EMBED * EMBED;

  const int nelem = ROWS * EMBED;
  convert_x_kernel<<<nelem / 256, 256, 0, stream>>>(x, xb, nelem);
  transpose_w_kernel<<<dim3(EMBED / 32, EMBED / 32, 4), dim3(32, 32), 0, stream>>>(
      Wq, Wk, Wv, Wo, Wt);

  const dim3 ggrid(ROWS / 256, EMBED / 64);
  gemm_bias_kernel<<<ggrid, 256, 0, stream>>>(xb, Wqt, bq, Qb, nullptr, 0);
  gemm_bias_kernel<<<ggrid, 256, 0, stream>>>(xb, Wkt, bk, Kb, nullptr, 1);
  gemm_bias_kernel<<<ggrid, 256, 0, stream>>>(xb, Wvt, bv, Vb, nullptr, 2);

  flash_attn_kernel<<<(BATCH * HEADS * (SEQ / 16)) / 4, 128, 0, stream>>>(
      Qb, Kb, Vb, xb);

  gemm_bias_kernel<<<ggrid, 256, 0, stream>>>(xb, Wot, bo, nullptr, out, 3);
}
